// Attention_5866925326364
// MI455X (gfx1250) — matrix operand order repaired and checked by reading, not compiled, not run on hardware
//
#include <hip/hip_runtime.h>

#define DDIM 512
#define SDIM 2048
#define BDIM 128
#define PITCH 520   // LDS row pitch (bf16 elems) = 1040B -> conflict-free A-tile reads
#define SBLK 128    // s-rows per workgroup
#define MT   2      // M-tiles (16 rows each) per wave -> 2x B reuse

typedef __attribute__((ext_vector_type(16))) __bf16 v16bf;
typedef __attribute__((ext_vector_type(8)))  __bf16 v8bf;
typedef __attribute__((ext_vector_type(8)))  float  v8f;
typedef __attribute__((ext_vector_type(4)))  int    i4v;

#define AS_GLOBAL __attribute__((address_space(1)))
#define AS_LDS    __attribute__((address_space(3)))

// gfx1250 async global->LDS path (guarded: falls back to sync copy if absent)
#if defined(__HIP_DEVICE_COMPILE__) && __has_builtin(__builtin_amdgcn_global_load_async_to_lds_b128)
#define HAVE_ASYNC_LDS 1
#else
#define HAVE_ASYNC_LDS 0
#endif

#if HAVE_ASYNC_LDS
#if __has_builtin(__builtin_amdgcn_s_wait_asynccnt)
#define WAIT_ASYNC() __builtin_amdgcn_s_wait_asynccnt(0)
#else
#define WAIT_ASYNC() asm volatile("s_wait_asynccnt 0x0" ::: "memory")
#endif
#else
#define WAIT_ASYNC()
#endif

// ---------------------------------------------------------------------------
// K0a: qbb[b,d] = (query @ Wq^T)[b,d] + bq[d] + br[d]
// ---------------------------------------------------------------------------
__global__ __launch_bounds__(256) void k_qproj(const float* __restrict__ query,
                                               const float* __restrict__ Wq,
                                               const float* __restrict__ bq,
                                               const float* __restrict__ br,
                                               float* __restrict__ qbb) {
    int idx = blockIdx.x * 256 + threadIdx.x;   // b*D + d
    int b = idx >> 9, d = idx & 511;
    const float4* qr = (const float4*)(query + (size_t)b * DDIM);
    const float4* wr = (const float4*)(Wq + (size_t)d * DDIM);
    float acc = 0.f;
#pragma unroll 4
    for (int k = 0; k < DDIM / 4; ++k) {
        float4 a = qr[k], w = wr[k];
        acc += a.x * w.x + a.y * w.y + a.z * w.z + a.w * w.w;
    }
    qbb[idx] = acc + bq[d] + br[d];
}

// ---------------------------------------------------------------------------
// K0b: repack Wr (f32 [n][k]) into bf16 WMMA-B tile blobs (1024B each).
// Tile (ntile 0..31, ktile 0..15); blob layout: elem (khalf, lane, j):
//   offset = khalf*256 + lane*8 + j   (bf16 elems), value = Wr[n][k] with
//   n = ntile*16 + (lane&15), k = ktile*32 + (lane>>4)*16 + khalf*8 + j.
// ---------------------------------------------------------------------------
__global__ __launch_bounds__(256) void k_wrtile(const float* __restrict__ Wr,
                                                __bf16* __restrict__ wrt) {
    int t = blockIdx.x * 256 + threadIdx.x;     // tile*32 + lane, 512 tiles
    int lane  = t & 31;
    int tile  = t >> 5;
    int ktile = tile & 15;
    int ntile = tile >> 4;
    int n  = ntile * 16 + (lane & 15);
    int kb = ktile * 32 + (lane >> 4) * 16;
    const float* src = Wr + (size_t)n * DDIM + kb;
    __bf16* dst = wrt + (size_t)tile * 512;
#pragma unroll
    for (int j = 0; j < 16; ++j)
        dst[(j >> 3) * 256 + lane * 8 + (j & 7)] = (__bf16)src[j];
}

// ---------------------------------------------------------------------------
// K1: logits u[b,s] = sum_d v[d] * tanh(qbb[b,d] + (Wr @ ref[s,b,:])[d])
// Workgroup = (b, 128-row s-block): ref slab -> LDS bf16 once.
// 8 waves split M and N: wave w owns rows (w&3)*32 (2 M-tiles) and N-half
// g=w>>2 (256 cols). Each N-half group double-buffers a 4 KB Wr panel in
// LDS (async global->LDS when available). Per-row u partials from the two
// halves are summed via LDS at the end.
// ---------------------------------------------------------------------------
__global__ __launch_bounds__(256) void k_scores(const float* __restrict__ ref,
                                                const __bf16* __restrict__ wrt,
                                                const float* __restrict__ qbb,
                                                const float* __restrict__ v,
                                                float* __restrict__ u_out) {
    __shared__ __bf16 refh[SBLK * PITCH];       // 130 KB
    __shared__ __bf16 bpanel[2][2][4 * 512];    // [group][buf] 4 KB panels
    __shared__ float  qb_s[DDIM];
    __shared__ float  v_s[DDIM];
    __shared__ float  upart[8][32];             // per-wave u partials

    const int b   = blockIdx.y;
    const int s0  = blockIdx.x * SBLK;
    const int tid = threadIdx.x;

    for (int i = tid; i < DDIM; i += 256) { qb_s[i] = qbb[(size_t)b * DDIM + i]; v_s[i] = v[i]; }

    // stage ref[s0..s0+127, b, :] -> bf16 LDS (coalesced float4 reads)
    for (int base = tid * 4; base < SBLK * DDIM; base += 1024) {
        int s = base >> 9, d = base & 511;
        float4 f = *(const float4*)(ref + ((size_t)(s0 + s) * BDIM + b) * DDIM + d);
        __bf16* p = &refh[s * PITCH + d];
        p[0] = (__bf16)f.x; p[1] = (__bf16)f.y; p[2] = (__bf16)f.z; p[3] = (__bf16)f.w;
    }

    const int lane  = tid & 31;
    const int w     = tid >> 5;
    const int g     = w >> 2;              // N-half this wave computes
    const int mbase = (w & 3) * 32;        // first s-row of this wave
    const int khi   = (lane >> 4) * 16;    // A-layout K sub-offset per half-wave

    // cooperative panel staging: thread's group / tile-in-panel / 32B chunk
    const int sg    = tid >> 7;            // staging group (matches wave group)
    const int t7    = tid & 127;
    const int ptile = t7 >> 5;             // 0..3
    const int pano  = (t7 & 31) * 2;       // int4 index inside 1 KB tile blob

    auto stage_panel = [&](int nc, int kk, int buf) {
        size_t tile = (size_t)((sg * 16 + nc * 4 + ptile) * 16 + kk);
#if HAVE_ASYNC_LDS
        const char* src = (const char*)wrt + tile * 1024 + (size_t)pano * 16;
        char*       dst = (char*)&bpanel[sg][buf][ptile * 512] + pano * 16;
        __builtin_amdgcn_global_load_async_to_lds_b128(
            (AS_GLOBAL i4v*)(void*)src, (AS_LDS i4v*)(void*)dst, 0, 0);
        __builtin_amdgcn_global_load_async_to_lds_b128(
            (AS_GLOBAL i4v*)(void*)(src + 16), (AS_LDS i4v*)(void*)(dst + 16), 0, 0);
#else
        const i4v* src = (const i4v*)wrt + tile * 64 + pano;
        i4v*       dst = (i4v*)&bpanel[sg][buf][ptile * 512] + pano;
        dst[0] = src[0];
        dst[1] = src[1];
#endif
    };

    float usum[MT][8];
#pragma unroll
    for (int mt = 0; mt < MT; ++mt)
#pragma unroll
        for (int r = 0; r < 8; ++r) usum[mt][r] = 0.f;

    for (int nc = 0; nc < 4; ++nc) {             // N chunk of 64 (per group)
        v8f acc[MT][4];
#pragma unroll
        for (int mt = 0; mt < MT; ++mt)
#pragma unroll
            for (int nt = 0; nt < 4; ++nt) acc[mt][nt] = (v8f)(0.f);

        stage_panel(nc, 0, 0);
        WAIT_ASYNC();
        __syncthreads();                         // also covers refh staging (nc==0)

        for (int kk = 0; kk < 16; ++kk) {        // K step of 32
            const int cur = kk & 1;
            if (kk < 15) stage_panel(nc, kk + 1, cur ^ 1);

            union { v16bf v; v8bf h[2]; } A[MT];
#pragma unroll
            for (int mt = 0; mt < MT; ++mt) {
                const __bf16* arow =
                    &refh[(mbase + mt * 16 + (lane & 15)) * PITCH + kk * 32 + khi];
                A[mt].h[0] = *(const v8bf*)(arow);
                A[mt].h[1] = *(const v8bf*)(arow + 8);
            }
#pragma unroll
            for (int nt = 0; nt < 4; ++nt) {
                union { v16bf v; v8bf h[2]; } Bm;
                Bm.h[0] = *(const v8bf*)&bpanel[g][cur][nt * 512 + lane * 8];
                Bm.h[1] = *(const v8bf*)&bpanel[g][cur][nt * 512 + 256 + lane * 8];
                acc[0][nt] = __builtin_amdgcn_wmma_f32_16x16x32_bf16(
                    false, A[0].v, false, Bm.v, (short)0, acc[0][nt], false, false);
                acc[1][nt] = __builtin_amdgcn_wmma_f32_16x16x32_bf16(
                    false, A[1].v, false, Bm.v, (short)0, acc[1][nt], false, false);
            }
            WAIT_ASYNC();
            __syncthreads();                     // next panel ready / safe to overwrite
        }

        // fused epilogue: usum += v[n] * tanh(qbb[n] + e)  (C layout: reg r<->M, lane<->N)
#pragma unroll
        for (int mt = 0; mt < MT; ++mt)
#pragma unroll
            for (int nt = 0; nt < 4; ++nt) {
                int n = g * 256 + nc * 64 + nt * 16 + (lane & 15);
                float qv = qb_s[n], vv = v_s[n];
#pragma unroll
                for (int r = 0; r < 8; ++r) {
                    float x  = qv + acc[mt][nt][r];
                    float ex = __expf(2.0f * x);                       // v_exp_f32
                    usum[mt][r] += vv * (1.0f - 2.0f * __builtin_amdgcn_rcpf(ex + 1.0f));
                }
            }
    }

    // reduce over N across each 16-lane half, park per-wave partials in LDS
#pragma unroll
    for (int mt = 0; mt < MT; ++mt)
#pragma unroll
        for (int r = 0; r < 8; ++r) {
            float s = usum[mt][r];
            s += __shfl_xor(s, 1, 16);
            s += __shfl_xor(s, 2, 16);
            s += __shfl_xor(s, 4, 16);
            s += __shfl_xor(s, 8, 16);
            if ((lane & 15) == 0)
                upart[w][mt * 16 + (lane >> 4) * 8 + r] = s;
        }
    __syncthreads();
    // combine the two N-half partials: row tid covered by waves (tid>>5, (tid>>5)+4)
    if (tid < SBLK)
        u_out[(size_t)b * SDIM + s0 + tid] =
            upart[tid >> 5][tid & 31] + upart[(tid >> 5) + 4][tid & 31];
}

// ---------------------------------------------------------------------------
// K2: in-place row softmax over S
// ---------------------------------------------------------------------------
__global__ __launch_bounds__(256) void k_softmax(float* __restrict__ scores) {
    __shared__ float red[256];
    __shared__ float row[SDIM];
    int b = blockIdx.x, tid = threadIdx.x;
    float* p = scores + (size_t)b * SDIM;
    float m = -3.4e38f;
    for (int i = tid; i < SDIM; i += 256) { float x = p[i]; row[i] = x; m = fmaxf(m, x); }
    red[tid] = m; __syncthreads();
    for (int st = 128; st > 0; st >>= 1) { if (tid < st) red[tid] = fmaxf(red[tid], red[tid + st]); __syncthreads(); }
    m = red[0]; __syncthreads();
    float sum = 0.f;
    for (int i = tid; i < SDIM; i += 256) { float e = __expf(row[i] - m); row[i] = e; sum += e; }
    red[tid] = sum; __syncthreads();
    for (int st = 128; st > 0; st >>= 1) { if (tid < st) red[tid] += red[tid + st]; __syncthreads(); }
    float inv = 1.0f / red[0];
    for (int i = tid; i < SDIM; i += 256) p[i] = row[i] * inv;
}

// ---------------------------------------------------------------------------
// K3: rbar[b,d] = sum_s scores[b,s] * ref[s,b,d]  (2nd and final ref pass)
// ---------------------------------------------------------------------------
__global__ __launch_bounds__(256) void k_rbar(const float* __restrict__ ref,
                                              const float* __restrict__ scores,
                                              float* __restrict__ rbar) {
    __shared__ float sc[SDIM];
    int b = blockIdx.x, tid = threadIdx.x;
    for (int i = tid; i < SDIM; i += 256) sc[i] = scores[(size_t)b * SDIM + i];
    __syncthreads();
    int d = blockIdx.y * 256 + tid;
    const float* rp = ref + (size_t)b * DDIM + d;
    const size_t stride = (size_t)BDIM * DDIM;
    float acc = 0.f;
    for (int s = 0; s < SDIM; s += 4) {
        int sp = s + 32; if (sp > SDIM - 1) sp = SDIM - 1;
        __builtin_prefetch(rp + (size_t)sp * stride, 0, 1);   // global_prefetch_b8
        acc += sc[s + 0] * rp[(size_t)(s + 0) * stride]
             + sc[s + 1] * rp[(size_t)(s + 1) * stride]
             + sc[s + 2] * rp[(size_t)(s + 2) * stride]
             + sc[s + 3] * rp[(size_t)(s + 3) * stride];
    }
    rbar[(size_t)b * DDIM + d] = acc;
}

// ---------------------------------------------------------------------------
// K4: att_state[b,:] = Wr @ rbar[b,:] + br   (scores sum to 1 absorbs br)
// ---------------------------------------------------------------------------
__global__ __launch_bounds__(256) void k_att(const float* __restrict__ Wr,
                                             const float* __restrict__ br,
                                             const float* __restrict__ rbar,
                                             float* __restrict__ att) {
    int idx = blockIdx.x * 256 + threadIdx.x;  // b*D + d
    int b = idx >> 9, d = idx & 511;
    const float4* wp = (const float4*)(Wr + (size_t)d * DDIM);
    const float4* rp = (const float4*)(rbar + (size_t)b * DDIM);
    float acc = 0.f;
#pragma unroll 4
    for (int k = 0; k < DDIM / 4; ++k) {
        float4 wv = wp[k], rv = rp[k];
        acc += wv.x * rv.x + wv.y * rv.y + wv.z * rv.z + wv.w * rv.w;
    }
    att[idx] = acc + br[d];
}

// ---------------------------------------------------------------------------
extern "C" void kernel_launch(void* const* d_in, const int* in_sizes, int n_in,
                              void* d_out, int out_size, void* d_ws, size_t ws_size,
                              hipStream_t stream) {
    const float* query = (const float*)d_in[0];
    const float* ref   = (const float*)d_in[1];
    const float* Wq    = (const float*)d_in[2];
    const float* bq    = (const float*)d_in[3];
    const float* Wr    = (const float*)d_in[4];
    const float* br    = (const float*)d_in[5];
    const float* v     = (const float*)d_in[6];

    float* out    = (float*)d_out;
    float* att    = out;                       // [B, D]
    float* scores = out + (size_t)BDIM * DDIM; // [B, S]

    char*   ws   = (char*)d_ws;
    __bf16* wrt  = (__bf16*)ws;                                  // 512 KB: Wr bf16 tile blobs
    float*  qbb  = (float*)(ws + 524288);                        // 256 KB: q + br
    float*  rbar = (float*)(ws + 524288 + 262144);               // 256 KB

    k_qproj  <<<dim3(BDIM * DDIM / 256), 256, 0, stream>>>(query, Wq, bq, br, qbb);
    k_wrtile <<<dim3(64),                256, 0, stream>>>(Wr, wrt);
    k_scores <<<dim3(SDIM / SBLK, BDIM), 256, 0, stream>>>(ref, wrt, qbb, v, scores);
    k_softmax<<<dim3(BDIM),              256, 0, stream>>>(scores);
    k_rbar   <<<dim3(BDIM, DDIM / 256),  256, 0, stream>>>(ref, scores, rbar);
    k_att    <<<dim3(BDIM * DDIM / 256), 256, 0, stream>>>(Wr, br, rbar, att);
}
